// GPT_24438363914268
// MI455X (gfx1250) — compile-verified, matrix operand order checked by reading, not run on hardware
//
#include <hip/hip_runtime.h>
#include <hip/hip_bf16.h>
#include <math.h>

// ---------------- model constants ----------------
#define Lnum 12
#define Hn   12
#define Cc   768
#define Dd   64
#define Tt   1024
#define Bb   4
#define Vv   50304
#define CM   3072                  // 4*C
#define Mrows (Bb*Tt)              // 4096 token rows

typedef __bf16 bf16_t;
typedef __attribute__((ext_vector_type(16))) __bf16 v16bf;
typedef __attribute__((ext_vector_type(8)))  float  v8f;

// ---------------- GEMM tiling ----------------
#define BM 128
#define BN 128
#define BK 32

static __device__ __forceinline__ v8f zero_v8f() {
    v8f z = {0.f,0.f,0.f,0.f,0.f,0.f,0.f,0.f};
    return z;
}

// CDNA5 async copy: global -> LDS, 16 bytes, tracked by ASYNCcnt.
// VDST = VGPR holding LDS byte address (low 32 bits of generic pointer:
// the LDS aperture occupies addr[63:32]); VADDR = 64-bit global address.
static __device__ __forceinline__ void async_copy16(void* lds_dst, const void* gsrc) {
    const unsigned int lds = (unsigned int)(unsigned long long)lds_dst;
    asm volatile("global_load_async_to_lds_b128 %0, %1, off"
                 :: "v"(lds), "v"(gsrc) : "memory");
}
static __device__ __forceinline__ void wait_async_all() {
    asm volatile("s_wait_asynccnt 0x0" ::: "memory");
}

// C[M,N] = A[M,K](bf16, row-major) * B[N,K](bf16, row-major)^T  (+ Res)
// Block: 256 threads = 8 waves, 128x128 tile. Wave grid 4(M) x 2(N);
// each wave owns 32x64 = 2 A-frags x 4 B-frags = 8 WMMAs per K=32 step.
// Double-buffered LDS fed by GLOBAL_LOAD_ASYNC_TO_LDS_B128.
__global__ __launch_bounds__(256)
void gemm_bf16_wmma(const bf16_t* __restrict__ A, const bf16_t* __restrict__ B,
                    float* __restrict__ Cout, const float* __restrict__ Res,
                    int M, int N, int K)
{
    __shared__ bf16_t sA[2][BM][BK];   // 2 x 8 KB
    __shared__ bf16_t sB[2][BN][BK];   // 2 x 8 KB

    const int tid  = threadIdx.x;
    const int lane = tid & 31;
    const int wave = tid >> 5;
    const int wm   = wave & 3;      // M sub-tile (0..3), 32 rows each
    const int wn   = wave >> 2;     // N sub-tile (0..1), 64 cols each
    const int l16  = lane & 15;
    const int half = lane >> 4;     // selects K-half of fragment
    const long bm  = (long)blockIdx.y * BM;
    const long bn  = (long)blockIdx.x * BN;

    // Stage one 128x32 A tile + 128x32 B tile: 512 chunks of 16B each,
    // 2 chunks/thread/matrix, all via async-to-LDS.
    auto stage = [&](int buf, int k0) {
        #pragma unroll
        for (int c = 0; c < 2; ++c) {
            const int chunk = tid + c * 256;       // 0..511
            const int row = chunk >> 2;
            const int col = (chunk & 3) * 8;       // 8 bf16 = 16B
            async_copy16(&sA[buf][row][col], &A[(size_t)(bm + row) * K + k0 + col]);
            async_copy16(&sB[buf][row][col], &B[(size_t)(bn + row) * K + k0 + col]);
        }
    };

    v8f acc[2][4];
    #pragma unroll
    for (int i = 0; i < 2; ++i)
        #pragma unroll
        for (int j = 0; j < 4; ++j) acc[i][j] = zero_v8f();

    stage(0, 0);
    int buf = 0;

    for (int k0 = 0; k0 < K; k0 += BK) {
        wait_async_all();          // current tile resident (this wave)
        __syncthreads();           // ... and all waves past previous reads

        // kick off next tile while we compute (overlaps with 8 WMMAs)
        if (k0 + BK < K) {
            stage(buf ^ 1, k0 + BK);
            // warm GL2 for the tile after next
            if (k0 + 2 * BK < K) {
                __builtin_prefetch(&A[(size_t)(bm + (tid >> 1)) * K + k0 + 2 * BK], 0, 1);
                __builtin_prefetch(&B[(size_t)(bn + (tid >> 1)) * K + k0 + 2 * BK], 0, 1);
            }
        }

        // fragments from LDS (32B per lane -> 2x ds_load_b128 each)
        const v16bf a0 = *(const v16bf*)&sA[buf][wm * 32 +  0 + l16][half * 16];
        const v16bf a1 = *(const v16bf*)&sA[buf][wm * 32 + 16 + l16][half * 16];
        const v16bf b0 = *(const v16bf*)&sB[buf][wn * 64 +  0 + l16][half * 16];
        const v16bf b1 = *(const v16bf*)&sB[buf][wn * 64 + 16 + l16][half * 16];
        const v16bf b2 = *(const v16bf*)&sB[buf][wn * 64 + 32 + l16][half * 16];
        const v16bf b3 = *(const v16bf*)&sB[buf][wn * 64 + 48 + l16][half * 16];

        acc[0][0] = __builtin_amdgcn_wmma_f32_16x16x32_bf16(false, a0, false, b0, (short)0, acc[0][0], false, false);
        acc[0][1] = __builtin_amdgcn_wmma_f32_16x16x32_bf16(false, a0, false, b1, (short)0, acc[0][1], false, false);
        acc[0][2] = __builtin_amdgcn_wmma_f32_16x16x32_bf16(false, a0, false, b2, (short)0, acc[0][2], false, false);
        acc[0][3] = __builtin_amdgcn_wmma_f32_16x16x32_bf16(false, a0, false, b3, (short)0, acc[0][3], false, false);
        acc[1][0] = __builtin_amdgcn_wmma_f32_16x16x32_bf16(false, a1, false, b0, (short)0, acc[1][0], false, false);
        acc[1][1] = __builtin_amdgcn_wmma_f32_16x16x32_bf16(false, a1, false, b1, (short)0, acc[1][1], false, false);
        acc[1][2] = __builtin_amdgcn_wmma_f32_16x16x32_bf16(false, a1, false, b2, (short)0, acc[1][2], false, false);
        acc[1][3] = __builtin_amdgcn_wmma_f32_16x16x32_bf16(false, a1, false, b3, (short)0, acc[1][3], false, false);

        buf ^= 1;
    }

    // epilogue: ISA C/D layout: VGPR i, lanes0-15 -> M=i, lanes16-31 -> M=i+8
    #pragma unroll
    for (int fi = 0; fi < 2; ++fi) {
        #pragma unroll
        for (int j = 0; j < 4; ++j) {
            const long col = bn + wn * 64 + j * 16 + l16;
            #pragma unroll
            for (int i = 0; i < 8; ++i) {
                const long row = bm + wm * 32 + fi * 16 + i + half * 8;
                const size_t o = (size_t)row * N + col;
                float v = acc[fi][j][i];
                if (Res) v += Res[o];
                Cout[o] = v;
            }
        }
    }
}

// ---------------- absmean ternary quantization ----------------
__global__ void zero_f32(float* p, int n) {
    if ((int)threadIdx.x < n) p[threadIdx.x] = 0.f;
}

__global__ __launch_bounds__(256)
void abs_sum(const float* __restrict__ w, float* __restrict__ gsum, size_t n)
{
    __shared__ float red[256];
    float s = 0.f;
    for (size_t i = (size_t)blockIdx.x * 256 + threadIdx.x; i < n;
         i += (size_t)gridDim.x * 256)
        s += fabsf(w[i]);
    red[threadIdx.x] = s;
    __syncthreads();
    for (int o = 128; o > 0; o >>= 1) {
        if ((int)threadIdx.x < o) red[threadIdx.x] += red[threadIdx.x + o];
        __syncthreads();
    }
    if (threadIdx.x == 0) atomicAdd(gsum, red[0]);
}

// q = sign(w) * min(round(|w|/gamma),1)  ==  (|w| < 0.5*gamma) ? 0 : sign(w)
__global__ __launch_bounds__(256)
void quant_ternary(const float* __restrict__ w, const float* __restrict__ gsum,
                   bf16_t* __restrict__ q, size_t n)
{
    const float gamma = (*gsum) / (float)n + 1e-6f;
    const float thr = 0.5f * gamma;
    for (size_t i = (size_t)blockIdx.x * 256 + threadIdx.x; i < n;
         i += (size_t)gridDim.x * 256) {
        const float v = w[i];
        const float t = (fabsf(v) < thr) ? 0.f : copysignf(1.f, v);
        q[i] = (bf16_t)t;
    }
}

__global__ __launch_bounds__(256)
void cast_f32_to_bf16(const float* __restrict__ src, bf16_t* __restrict__ dst, size_t n)
{
    for (size_t i = (size_t)blockIdx.x * 256 + threadIdx.x; i < n;
         i += (size_t)gridDim.x * 256)
        dst[i] = (bf16_t)src[i];
}

// ---------------- embedding ----------------
__global__ __launch_bounds__(256)
void embed_kernel(const int* __restrict__ idx, const float* __restrict__ wte,
                  const float* __restrict__ wpe, float* __restrict__ x)
{
    const int i = blockIdx.x * 256 + threadIdx.x;
    if (i >= Mrows * Cc) return;
    const int r = i / Cc, c = i - r * Cc;
    const int tpos = r & (Tt - 1);
    x[i] = wte[(size_t)idx[r] * Cc + c] + wpe[(size_t)tpos * Cc + c];
}

// ---------------- layernorm (f32 in -> bf16 out) ----------------
__global__ __launch_bounds__(256)
void layernorm_bf16(const float* __restrict__ x, const float* __restrict__ w,
                    const float* __restrict__ b, bf16_t* __restrict__ out, int width)
{
    __shared__ float red[256];
    const int row = blockIdx.x;
    const float* xr = x + (size_t)row * width;

    float s = 0.f;
    for (int i = threadIdx.x; i < width; i += 256) s += xr[i];
    red[threadIdx.x] = s; __syncthreads();
    for (int o = 128; o > 0; o >>= 1) {
        if ((int)threadIdx.x < o) red[threadIdx.x] += red[threadIdx.x + o];
        __syncthreads();
    }
    const float mu = red[0] / width;
    __syncthreads();

    float v = 0.f;
    for (int i = threadIdx.x; i < width; i += 256) { const float d = xr[i] - mu; v += d * d; }
    red[threadIdx.x] = v; __syncthreads();
    for (int o = 128; o > 0; o >>= 1) {
        if ((int)threadIdx.x < o) red[threadIdx.x] += red[threadIdx.x + o];
        __syncthreads();
    }
    const float rstd = rsqrtf(red[0] / width + 1e-5f);

    bf16_t* orow = out + (size_t)row * width;
    for (int i = threadIdx.x; i < width; i += 256)
        orow[i] = (bf16_t)((xr[i] - mu) * rstd * w[i] + b[i]);
}

// ---------------- attention: online softmax, 1 thread / query row ----------------
__global__ __launch_bounds__(256)
void attention_kernel(const float* __restrict__ qkv, bf16_t* __restrict__ y)
{
    const int t = blockIdx.x * 256 + threadIdx.x;
    if (t >= Bb * Hn * Tt) return;
    const int q = t & (Tt - 1);
    const int h = (t / Tt) % Hn;
    const int b = t / (Tt * Hn);

    const size_t rowstride = (size_t)3 * Cc;
    const float* qrow = qkv + ((size_t)(b * Tt + q)) * rowstride + h * Dd;

    float4 qv[Dd / 4];
    #pragma unroll
    for (int i = 0; i < Dd / 4; ++i) qv[i] = *(const float4*)(qrow + i * 4);

    float4 acc[Dd / 4];
    #pragma unroll
    for (int i = 0; i < Dd / 4; ++i) acc[i] = make_float4(0.f, 0.f, 0.f, 0.f);

    float m = -3.402823466e38f, l = 0.f;
    const float scale = 0.125f;  // 1/sqrt(64)

    for (int k = 0; k <= q; ++k) {
        const float* krow = qkv + ((size_t)(b * Tt + k)) * rowstride + Cc + h * Dd;
        float s = 0.f;
        #pragma unroll
        for (int i = 0; i < Dd / 4; ++i) {
            const float4 kv = *(const float4*)(krow + i * 4);
            s += qv[i].x * kv.x + qv[i].y * kv.y + qv[i].z * kv.z + qv[i].w * kv.w;
        }
        s *= scale;
        const float nm = fmaxf(m, s);
        const float coef = __expf(m - nm);
        const float p = __expf(s - nm);
        l = l * coef + p;
        const float* vrow = qkv + ((size_t)(b * Tt + k)) * rowstride + 2 * Cc + h * Dd;
        #pragma unroll
        for (int i = 0; i < Dd / 4; ++i) {
            const float4 vv = *(const float4*)(vrow + i * 4);
            acc[i].x = acc[i].x * coef + p * vv.x;
            acc[i].y = acc[i].y * coef + p * vv.y;
            acc[i].z = acc[i].z * coef + p * vv.z;
            acc[i].w = acc[i].w * coef + p * vv.w;
        }
        m = nm;
    }
    const float inv = 1.f / l;
    bf16_t* yrow = y + ((size_t)(b * Tt + q)) * Cc + h * Dd;
    #pragma unroll
    for (int i = 0; i < Dd / 4; ++i) {
        yrow[i * 4 + 0] = (bf16_t)(acc[i].x * inv);
        yrow[i * 4 + 1] = (bf16_t)(acc[i].y * inv);
        yrow[i * 4 + 2] = (bf16_t)(acc[i].z * inv);
        yrow[i * 4 + 3] = (bf16_t)(acc[i].w * inv);
    }
}

// ---------------- exact GELU ----------------
__global__ __launch_bounds__(256)
void gelu_kernel(float* __restrict__ x, size_t n)
{
    for (size_t i = (size_t)blockIdx.x * 256 + threadIdx.x; i < n;
         i += (size_t)gridDim.x * 256) {
        const float v = x[i];
        x[i] = 0.5f * v * (1.f + erff(v * 0.70710678118654752f));
    }
}

// ---------------- host orchestration ----------------
extern "C" void kernel_launch(void* const* d_in, const int* in_sizes, int n_in,
                              void* d_out, int out_size, void* d_ws, size_t ws_size,
                              hipStream_t stream)
{
    const int*   idx      = (const int*)  d_in[0];
    const float* wte      = (const float*)d_in[1];
    const float* wpe      = (const float*)d_in[2];
    const float* ln1_w    = (const float*)d_in[3];
    const float* ln1_b    = (const float*)d_in[4];
    const float* attn_W   = (const float*)d_in[5];
    const float* proj_W   = (const float*)d_in[6];
    const float* ln2_w    = (const float*)d_in[7];
    const float* ln2_b    = (const float*)d_in[8];
    const float* fc_W     = (const float*)d_in[9];
    const float* mlp_ln_w = (const float*)d_in[10];
    const float* mlp_ln_b = (const float*)d_in[11];
    const float* proj2_W  = (const float*)d_in[12];
    const float* lnf_w    = (const float*)d_in[13];
    const float* lnf_b    = (const float*)d_in[14];
    float* out = (float*)d_out;

    // ---- carve workspace ----
    char* base = (char*)d_ws;
    auto carve = [&](size_t bytes) -> char* {
        char* p = base;
        base += (bytes + 255) & ~(size_t)255;
        return p;
    };
    float*  x    = (float*) carve((size_t)Mrows * Cc * 4);       // residual stream
    bf16_t* h    = (bf16_t*)carve((size_t)Mrows * CM * 2);       // ln outputs (<=3072 wide)
    float*  qkv  = (float*) carve((size_t)Mrows * 3 * Cc * 4);
    bf16_t* y    = (bf16_t*)carve((size_t)Mrows * Cc * 2);
    float*  big  = (float*) carve((size_t)Mrows * CM * 4);       // fc output
    bf16_t* qa   = (bf16_t*)carve((size_t)3 * Cc * Cc * 2);
    bf16_t* qp   = (bf16_t*)carve((size_t)Cc * Cc * 2);
    bf16_t* qf   = (bf16_t*)carve((size_t)CM * Cc * 2);
    bf16_t* q2   = (bf16_t*)carve((size_t)Cc * CM * 2);
    bf16_t* wteb = (bf16_t*)carve((size_t)Vv * Cc * 2);
    float*  gam  = (float*) carve(64);

    const dim3 blk(256);

    // lm_head weights in bf16
    cast_f32_to_bf16<<<dim3(4096), blk, 0, stream>>>(wte, wteb, (size_t)Vv * Cc);

    // token + position embedding
    embed_kernel<<<dim3((Mrows * Cc + 255) / 256), blk, 0, stream>>>(idx, wte, wpe, x);

    const size_t nA = (size_t)3 * Cc * Cc;   // 2304*768
    const size_t nP = (size_t)Cc * Cc;       // 768*768
    const size_t nF = (size_t)CM * Cc;       // 3072*768
    const size_t n2 = (size_t)Cc * CM;       // 768*3072

    for (int l = 0; l < Lnum; ++l) {
        const float* aW  = attn_W  + (size_t)l * nA;
        const float* pW  = proj_W  + (size_t)l * nP;
        const float* fW  = fc_W    + (size_t)l * nF;
        const float* p2W = proj2_W + (size_t)l * n2;

        // ---- absmean ternary quantization of this layer's weights ----
        zero_f32<<<1, 32, 0, stream>>>(gam, 4);
        abs_sum<<<dim3(1024), blk, 0, stream>>>(aW,  gam + 0, nA);
        abs_sum<<<dim3(1024), blk, 0, stream>>>(pW,  gam + 1, nP);
        abs_sum<<<dim3(1024), blk, 0, stream>>>(fW,  gam + 2, nF);
        abs_sum<<<dim3(1024), blk, 0, stream>>>(p2W, gam + 3, n2);
        quant_ternary<<<dim3(2048), blk, 0, stream>>>(aW,  gam + 0, qa, nA);
        quant_ternary<<<dim3(2048), blk, 0, stream>>>(pW,  gam + 1, qp, nP);
        quant_ternary<<<dim3(2048), blk, 0, stream>>>(fW,  gam + 2, qf, nF);
        quant_ternary<<<dim3(2048), blk, 0, stream>>>(p2W, gam + 3, q2, n2);

        // ---- attention ----
        layernorm_bf16<<<dim3(Mrows), blk, 0, stream>>>(x, ln1_w + (size_t)l * Cc,
                                                        ln1_b + (size_t)l * Cc, h, Cc);
        gemm_bf16_wmma<<<dim3(3 * Cc / BN, Mrows / BM), blk, 0, stream>>>(
            h, qa, qkv, nullptr, Mrows, 3 * Cc, Cc);
        attention_kernel<<<dim3((Bb * Hn * Tt + 255) / 256), blk, 0, stream>>>(qkv, y);
        gemm_bf16_wmma<<<dim3(Cc / BN, Mrows / BM), blk, 0, stream>>>(
            y, qp, x, x, Mrows, Cc, Cc);

        // ---- MLP ----
        layernorm_bf16<<<dim3(Mrows), blk, 0, stream>>>(x, ln2_w + (size_t)l * Cc,
                                                        ln2_b + (size_t)l * Cc, h, Cc);
        gemm_bf16_wmma<<<dim3(CM / BN, Mrows / BM), blk, 0, stream>>>(
            h, qf, big, nullptr, Mrows, CM, Cc);
        gelu_kernel<<<dim3(4096), blk, 0, stream>>>(big, (size_t)Mrows * CM);
        layernorm_bf16<<<dim3(Mrows), blk, 0, stream>>>(big, mlp_ln_w + (size_t)l * CM,
                                                        mlp_ln_b + (size_t)l * CM, h, CM);
        gemm_bf16_wmma<<<dim3(Cc / BN, Mrows / BM), blk, 0, stream>>>(
            h, q2, x, x, Mrows, Cc, CM);
    }

    // ---- final LN + tied lm_head ----
    layernorm_bf16<<<dim3(Mrows), blk, 0, stream>>>(x, lnf_w, lnf_b, h, Cc);
    gemm_bf16_wmma<<<dim3(Vv / BN, Mrows / BM), blk, 0, stream>>>(
        h, wteb, out, nullptr, Mrows, Vv, Cc);
}
